// UVNetGraphEncoder_No_Edge_7310034338048
// MI455X (gfx1250) — compile-verified, matrix operand order checked by reading, not run on hardware
//
#include <hip/hip_runtime.h>
#include <hip/hip_bf16.h>

#define NN 50000
#define EE 600000
#define DD 128
#define BB 8
#define LRELU 0.01f

typedef __bf16 bf16_t;
typedef __attribute__((ext_vector_type(16))) __bf16 v16bf;
typedef __attribute__((ext_vector_type(8)))  float  v8f;

// ---------- helpers ----------
__device__ __forceinline__ unsigned f2key(float f) {
    unsigned u = __float_as_uint(f);
    return (u >> 31) ? ~u : (u | 0x80000000u);
}
__device__ __forceinline__ float key2f(unsigned k) {
    unsigned u = (k >> 31) ? (k & 0x7fffffffu) : ~k;
    return __uint_as_float(u);
}

// ---------- zero fill ----------
__global__ void k_zero(float* p, long n) {
    long i = (long)blockIdx.x * blockDim.x + threadIdx.x;
    if (i < n) p[i] = 0.0f;
}

// ---------- edge scatter: agg[dst] += x[src] ----------
__global__ void k_scatter(const float* __restrict__ x,
                          const int* __restrict__ src,
                          const int* __restrict__ dst,
                          float* __restrict__ agg) {
    long idx = (long)blockIdx.x * blockDim.x + threadIdx.x;
    if (idx >= (long)EE * 32) return;
    int e = (int)(idx >> 5);
    int g = (int)(idx & 31);          // 4-float group within the 128-dim row
    int s = src[e], d = dst[e];
    const float4 v = *(const float4*)(x + (long)s * DD + g * 4);
    float* o = agg + (long)d * DD + g * 4;
    atomicAdd(o + 0, v.x);
    atomicAdd(o + 1, v.y);
    atomicAdd(o + 2, v.z);
    atomicAdd(o + 3, v.w);
}

// ---------- fused-A bf16 WMMA GEMM: Y = A(fused) * W + bias ----------
// mode 0: A = (1+eps)*X + AGG               (GIN combine)
// mode 1: A = relu(scale[k]*X + shift[k])   (BN + ReLU of previous GEMM)
__global__ __launch_bounds__(256)
void k_gemm_wmma(int mode,
                 const float* __restrict__ X,
                 const float* __restrict__ AGG,
                 const float* __restrict__ epsP,
                 const float* __restrict__ scale,
                 const float* __restrict__ shift,
                 const float* __restrict__ W,      // [128][128] row-major (k, n)
                 const float* __restrict__ bias,   // [128]
                 float* __restrict__ Y,
                 int nrows) {
    // W staged in LDS pre-swizzled into WMMA B-fragment order:
    // [kc(4)][tile(8)][lane(32)][elem(16)]  -> each fragment is one aligned
    // 32-byte contiguous run => 2x ds_load_b128 per fragment.
    __shared__ __attribute__((aligned(32))) bf16_t Wswz[DD * DD];
    for (int i = threadIdx.x; i < DD * DD; i += 256) {
        int e  = i & 15;
        int ln = (i >> 4) & 31;
        int t  = (i >> 9) & 7;
        int kc = i >> 12;
        int n  = t * 16 + (ln & 15);
        int kb = kc * 32 + (ln >> 4) * 8;
        int j  = e >> 1, lo = e & 1;
        int k  = (j < 4) ? (kb + 2 * j + lo) : (kb + 16 + 2 * (j - 4) + lo);
        Wswz[i] = (bf16_t)W[k * DD + n];
    }
    __syncthreads();

    const int wave = threadIdx.x >> 5;
    const int lane = threadIdx.x & 31;
    const int n16  = lane & 15;
    const int hi   = lane >> 4;                    // half-wave select
    const long rowStrip = (long)blockIdx.x * 128 + wave * 16;
    long r = rowStrip + n16;                       // A-row owned by this lane
    const bool rin = (r < nrows);
    if (!rin) r = 0;                               // clamp: garbage row never stored
    const float eps1 = (mode == 0) ? (1.0f + epsP[0]) : 1.0f;
    const float* __restrict__ xr = X + r * DD;
    const float* __restrict__ gr = AGG ? (AGG + r * DD) : X;  // only read in mode 0

    v8f acc[8] = {};                               // 8 N-tiles of 16x16 f32

    #pragma unroll
    for (int kc = 0; kc < 4; ++kc) {               // K = 128 in chunks of 32
        const int kb = kc * 32 + hi * 8;
        // A fragment elems: e<8 -> K = kb+e ; e>=8 -> K = kb+16+(e-8)
        float xv[16];
        *(float4*)(xv + 0)  = *(const float4*)(xr + kb);
        *(float4*)(xv + 4)  = *(const float4*)(xr + kb + 4);
        *(float4*)(xv + 8)  = *(const float4*)(xr + kb + 16);
        *(float4*)(xv + 12) = *(const float4*)(xr + kb + 20);

        v16bf af;
        if (mode == 0) {
            float gv[16];
            *(float4*)(gv + 0)  = *(const float4*)(gr + kb);
            *(float4*)(gv + 4)  = *(const float4*)(gr + kb + 4);
            *(float4*)(gv + 8)  = *(const float4*)(gr + kb + 16);
            *(float4*)(gv + 12) = *(const float4*)(gr + kb + 20);
            #pragma unroll
            for (int e = 0; e < 16; ++e)
                af[e] = (bf16_t)(eps1 * xv[e] + gv[e]);
        } else {
            float sc[16], sf[16];
            *(float4*)(sc + 0)  = *(const float4*)(scale + kb);
            *(float4*)(sc + 4)  = *(const float4*)(scale + kb + 4);
            *(float4*)(sc + 8)  = *(const float4*)(scale + kb + 16);
            *(float4*)(sc + 12) = *(const float4*)(scale + kb + 20);
            *(float4*)(sf + 0)  = *(const float4*)(shift + kb);
            *(float4*)(sf + 4)  = *(const float4*)(shift + kb + 4);
            *(float4*)(sf + 8)  = *(const float4*)(shift + kb + 16);
            *(float4*)(sf + 12) = *(const float4*)(shift + kb + 20);
            #pragma unroll
            for (int e = 0; e < 16; ++e) {
                float v = sc[e] * xv[e] + sf[e];
                af[e] = (bf16_t)(v > 0.f ? v : 0.f);
            }
        }

        #pragma unroll
        for (int t = 0; t < 8; ++t) {
            const v16bf bfr =
                *(const v16bf*)(Wswz + (((kc * 8 + t) * 32 + lane) << 4));
            acc[t] = __builtin_amdgcn_wmma_f32_16x16x32_bf16(
                false, af, false, bfr, (short)0, acc[t], false, false);
        }
    }

    // store D (f32 16x16 layout: n = lane&15, m = v + 8*hi)
    #pragma unroll
    for (int t = 0; t < 8; ++t) {
        const int col = t * 16 + n16;
        const float bb = bias[col];
        #pragma unroll
        for (int v = 0; v < 8; ++v) {
            long rr = rowStrip + v + 8 * hi;
            if (rr < nrows) Y[rr * DD + col] = acc[t][v] + bb;
        }
    }
}

// ---------- per-column sum / sumsq (BN batch stats) ----------
__global__ __launch_bounds__(256)
void k_col_stats(const float* __restrict__ Y, float* __restrict__ sums,
                 float* __restrict__ sumsq, int nrows) {
    const int col  = threadIdx.x & 127;
    const int half = threadIdx.x >> 7;
    long r0   = (long)blockIdx.x * 256 + half;
    long rend = (long)blockIdx.x * 256 + 256;
    if (rend > nrows) rend = nrows;
    float s = 0.f, s2 = 0.f;
    for (long r = r0; r < rend; r += 2) {
        float v = Y[r * DD + col];
        s += v; s2 += v * v;
    }
    __shared__ float sh[256], sh2[256];
    sh[threadIdx.x] = s; sh2[threadIdx.x] = s2;
    __syncthreads();
    if (half == 0) {
        atomicAdd(&sums[col],  s  + sh[col + 128]);
        atomicAdd(&sumsq[col], s2 + sh2[col + 128]);
    }
}

__global__ void k_bn_finalize(const float* sums, const float* sumsq,
                              const float* g, const float* be,
                              float* scale, float* shift, float invN) {
    int c = threadIdx.x;
    float mean = sums[c] * invN;
    float var  = sumsq[c] * invN - mean * mean;
    float inv  = rsqrtf(var + 1e-5f);
    float sc   = g[c] * inv;
    scale[c] = sc;
    shift[c] = be[c] - mean * sc;
}

// ---------- final BN + leaky ReLU ----------
__global__ void k_bn_leaky(const float* __restrict__ Y, const float* __restrict__ scale,
                           const float* __restrict__ shift, float* __restrict__ out, long n) {
    long i = (long)blockIdx.x * blockDim.x + threadIdx.x;
    if (i >= n) return;
    int c = (int)(i & 127);
    float t = scale[c] * Y[i] + shift[c];
    out[i] = t > 0.f ? t : LRELU * t;
}

// ---------- segment max pooling (rows sorted by segment) ----------
#define POOLR 512
__global__ __launch_bounds__(128)
void k_pool_max(const float* __restrict__ H, const int* __restrict__ seg,
                unsigned* __restrict__ pooled) {
    const int col = threadIdx.x;
    long r0   = (long)blockIdx.x * POOLR;
    long rend = r0 + POOLR; if (rend > NN) rend = NN;
    int cur = -1; unsigned mk = 0u;
    for (long r = r0; r < rend; ++r) {
        int s = seg[r];
        if (s != cur) {
            if (cur >= 0) atomicMax(&pooled[cur * DD + col], mk);
            cur = s; mk = 0u;
        }
        unsigned key = f2key(H[r * DD + col]);
        mk = mk > key ? mk : key;
    }
    if (cur >= 0) atomicMax(&pooled[cur * DD + col], mk);
}

// ---------- score = sum_i pooled_i @ W_i + b_i ----------
__global__ __launch_bounds__(128)
void k_score(const unsigned* p0, const unsigned* p1, const unsigned* p2,
             const float* W0, const float* b0,
             const float* W1, const float* b1,
             const float* W2, const float* b2,
             float* out) {
    int o = threadIdx.x;
    for (int b = 0; b < BB; ++b) {
        float acc = b0[o] + b1[o] + b2[o];
        for (int k = 0; k < DD; ++k) {
            acc += key2f(p0[b * DD + k]) * W0[k * DD + o];
            acc += key2f(p1[b * DD + k]) * W1[k * DD + o];
            acc += key2f(p2[b * DD + k]) * W2[k * DD + o];
        }
        out[b * DD + o] = acc;
    }
}

extern "C" void kernel_launch(void* const* d_in, const int* in_sizes, int n_in,
                              void* d_out, int out_size, void* d_ws, size_t ws_size,
                              hipStream_t stream) {
    const float* h   = (const float*)d_in[0];
    const int*   src = (const int*)d_in[1];
    const int*   dst = (const int*)d_in[2];
    const int*   seg = (const int*)d_in[3];
    // conv params: [eps, W1, b1, g1, be1, W2, b2, g2, be2] per layer
    const float* c_eps[2] = { (const float*)d_in[4],  (const float*)d_in[13] };
    const float* c_W1[2]  = { (const float*)d_in[5],  (const float*)d_in[14] };
    const float* c_b1[2]  = { (const float*)d_in[6],  (const float*)d_in[15] };
    const float* c_g1[2]  = { (const float*)d_in[7],  (const float*)d_in[16] };
    const float* c_be1[2] = { (const float*)d_in[8],  (const float*)d_in[17] };
    const float* c_W2[2]  = { (const float*)d_in[9],  (const float*)d_in[18] };
    const float* c_b2[2]  = { (const float*)d_in[10], (const float*)d_in[19] };
    const float* c_g2[2]  = { (const float*)d_in[11], (const float*)d_in[20] };
    const float* c_be2[2] = { (const float*)d_in[12], (const float*)d_in[21] };
    const float* p_W[3]   = { (const float*)d_in[22], (const float*)d_in[24], (const float*)d_in[26] };
    const float* p_b[3]   = { (const float*)d_in[23], (const float*)d_in[25], (const float*)d_in[27] };

    float* out_h2    = (float*)d_out;                 // N*128
    float* out_score = (float*)d_out + (long)NN * DD; // 8*128

    // workspace layout
    float* bufA = (float*)d_ws;                       // agg, then GEMM2 output
    float* bufB = bufA + (long)NN * DD;               // GEMM1 output
    float* h1   = bufB + (long)NN * DD;               // layer-0 result
    float* sums  = h1 + (long)NN * DD;                // 128
    float* sumsq = sums + DD;                         // 128
    float* scale = sumsq + DD;                        // 128
    float* shift = scale + DD;                        // 128
    unsigned* pooled = (unsigned*)(shift + DD);       // 3 * 8 * 128

    const long nElem   = (long)NN * DD;
    const int  zBlk    = (int)((nElem + 255) / 256);
    const int  sBlk    = (int)(((long)EE * 32 + 255) / 256);
    const int  gBlk    = (NN + 127) / 128;
    const int  statBlk = (NN + 255) / 256;
    const int  poolBlk = (NN + POOLR - 1) / POOLR;
    const float invN   = 1.0f / (float)NN;

    const float* layer_in[2]  = { h, h1 };
    float*       layer_out[2] = { h1, out_h2 };

    for (int l = 0; l < 2; ++l) {
        // 1) aggregation
        k_zero<<<zBlk, 256, 0, stream>>>(bufA, nElem);
        k_scatter<<<sBlk, 256, 0, stream>>>(layer_in[l], src, dst, bufA);
        // 2) GEMM1: y = ((1+eps)*x + agg) @ W1 + b1
        k_gemm_wmma<<<gBlk, 256, 0, stream>>>(0, layer_in[l], bufA, c_eps[l],
                                              scale, shift,
                                              c_W1[l], c_b1[l], bufB, NN);
        // 3) BN1 stats
        k_zero<<<1, 256, 0, stream>>>(sums, 256);     // sums+sumsq contiguous
        k_col_stats<<<statBlk, 256, 0, stream>>>(bufB, sums, sumsq, NN);
        k_bn_finalize<<<1, DD, 0, stream>>>(sums, sumsq, c_g1[l], c_be1[l], scale, shift, invN);
        // 4) GEMM2: y2 = relu(bn(y)) @ W2 + b2
        k_gemm_wmma<<<gBlk, 256, 0, stream>>>(1, bufB, nullptr, c_eps[l],
                                              scale, shift,
                                              c_W2[l], c_b2[l], bufA, NN);
        // 5) BN2 stats + leaky ReLU
        k_zero<<<1, 256, 0, stream>>>(sums, 256);
        k_col_stats<<<statBlk, 256, 0, stream>>>(bufA, sums, sumsq, NN);
        k_bn_finalize<<<1, DD, 0, stream>>>(sums, sumsq, c_g2[l], c_be2[l], scale, shift, invN);
        k_bn_leaky<<<zBlk, 256, 0, stream>>>(bufA, scale, shift, layer_out[l], nElem);
    }

    // pooling + score
    k_zero<<<12, 256, 0, stream>>>((float*)pooled, 3 * BB * DD);  // key 0 == -inf
    k_pool_max<<<poolBlk, DD, 0, stream>>>(h,      seg, pooled);
    k_pool_max<<<poolBlk, DD, 0, stream>>>(h1,     seg, pooled + BB * DD);
    k_pool_max<<<poolBlk, DD, 0, stream>>>(out_h2, seg, pooled + 2 * BB * DD);
    k_score<<<1, DD, 0, stream>>>(pooled, pooled + BB * DD, pooled + 2 * BB * DD,
                                  p_W[0], p_b[0], p_W[1], p_b[1], p_W[2], p_b[2],
                                  out_score);
}
